// PyTorchRNNDecoder_82016695484934
// MI455X (gfx1250) — matrix operand order repaired and checked by reading, not compiled, not run on hardware
//
#include <hip/hip_runtime.h>
#include <hip/hip_bf16.h>
#include <cstdint>

// ---------------------------------------------------------------------------
// RNN-T decoder for MI455X (gfx1250, wave32, WMMA + async LDS staging).
// Dominant cost: joint GEMM [81600 x 1024 x 512] -> bf16 WMMA, fp32 accum.
// ---------------------------------------------------------------------------

typedef __attribute__((ext_vector_type(16))) __bf16 v16bf;
typedef __attribute__((ext_vector_type(8)))  float  v8f;
typedef __attribute__((ext_vector_type(4)))  unsigned int u32x4;

union ABfrag { u32x4 q[2]; v16bf v; };

#define N_B   8
#define T_I   200
#define T_O1  51
#define V_V   1024
#define E_E   512
#define H_H   512
#define J_J   512
#define L_L   3

// ---- gfx1250 async global->LDS copy (ASYNCcnt) with portable fallback -----
#if defined(__has_builtin)
#if __has_builtin(__builtin_amdgcn_global_load_async_to_lds_b128)
#define HAVE_ASYNC 1
#endif
#endif
#ifndef HAVE_ASYNC
#define HAVE_ASYNC 0
#endif

typedef int v4i __attribute__((vector_size(16)));
typedef __attribute__((address_space(1))) v4i as1_v4i;   // global (__device__)
typedef __attribute__((address_space(3))) v4i as3_v4i;   // LDS (__shared__)

static __device__ __forceinline__ void async_cp16(const __bf16* g, __bf16* l)
{
#if HAVE_ASYNC
    __builtin_amdgcn_global_load_async_to_lds_b128(
        (as1_v4i*)(uintptr_t)g,
        (as3_v4i*)(unsigned int)(uintptr_t)l,
        0, 0);
#else
    *(u32x4*)l = *(const u32x4*)g;
#endif
}

template <int Ncnt>
static __device__ __forceinline__ void wait_async()
{
#if HAVE_ASYNC
#if __has_builtin(__builtin_amdgcn_s_wait_asynccnt)
    __builtin_amdgcn_s_wait_asynccnt(Ncnt);
#else
    asm volatile("s_wait_asynccnt %0" :: "i"(Ncnt));
#endif
#endif
}

static __device__ __forceinline__ float sigm(float x) {
    return 1.0f / (1.0f + __expf(-x));
}

// ---------------------------------------------------------------------------
// Generic bf16 WMMA GEMM: C[M,N] = A[M,K] * W[N,K]^T (+bias[N])
// Requirements: K % 64 == 0, N % 128 == 0. M guarded.
// Block: 256 threads (8 waves), tile 128x128, waves in 4(M) x 2(N) grid,
// each wave owns 2x4 tiles of 16x16.
// A is staged into LDS in 128x64 chunks via async copies (double-buffered,
// 144B padded row stride -> conflict-free ds_load_b128 fragment reads).
// ---------------------------------------------------------------------------
#define KCH 64           // K chunk staged per buffer
#define AROWB 72         // bf16 elements per LDS row (144 bytes)

__global__ __launch_bounds__(256) void gemm_bf16_kernel(
    const __bf16* __restrict__ A, const __bf16* __restrict__ W,
    const float* __restrict__ bias, float* __restrict__ C,
    int M, int N, int K)
{
    __shared__ __bf16 ash[2][128 * AROWB];   // 2 x 18 KB

    const int tid   = threadIdx.x;
    const int lane  = tid & 31;
    const int wave  = tid >> 5;
    const int waveM = wave >> 1;          // 0..3
    const int waveN = wave & 1;           // 0..1
    const int l15   = lane & 15;

    const int rowBlock = blockIdx.x * 128;
    const int colTile0 = blockIdx.y * 128 + waveN * 64;  // + tc*16

    // A fragment (16x32): lane<16 -> K=k+0..7 / k+16..23 ; lane>=16 -> +8
    const int akoff = (lane < 16) ? 0 : 8;
    // B fragment (32x16): lane<16 -> K=k+0..7 / k+16..23 ; lane>=16 -> +8
    const int bkoff = (lane < 16) ? 0 : 8;

    // async chunk copy: 128 rows x 128B; 256 threads x 4 iters x 16B
    auto issue_chunk = [&](int k0, int buf) {
        for (int it = 0; it < 4; ++it) {
            const int f  = tid * 16 + it * 4096;  // byte index in chunk
            const int r  = f >> 7;                // row 0..127
            const int bo = f & 127;               // byte within row
            int row = rowBlock + r;
            if (row >= M) row = M - 1;            // clamp; stores guarded
            const __bf16* g = A + (size_t)row * K + k0 + (bo >> 1);
            __bf16* l = &ash[buf][r * AROWB + (bo >> 1)];
            async_cp16(g, l);
        }
    };

    v8f acc[2][4] = {};

    const __bf16* bptr[4];
    for (int tc = 0; tc < 4; ++tc) {
        const int cL = colTile0 + tc * 16 + l15;
        bptr[tc] = W + (size_t)cL * K + bkoff;
    }

    issue_chunk(0, 0);
    for (int kc = 0; kc < K; kc += KCH) {
        const int  buf  = (kc / KCH) & 1;
        const bool more = (kc + KCH) < K;
        if (more) {
            issue_chunk(kc + KCH, buf ^ 1);
            wait_async<4>();        // current-buffer group (4 issues) landed
        } else {
            wait_async<0>();
        }
        __syncthreads();            // all waves' copies visible

        for (int kk = 0; kk < KCH; kk += 32) {
            ABfrag a[2], b[4];
            for (int tr = 0; tr < 2; ++tr) {
                const __bf16* ap =
                    &ash[buf][(waveM * 32 + tr * 16 + l15) * AROWB + kk + akoff];
                a[tr].q[0] = *(const u32x4*)(ap);
                a[tr].q[1] = *(const u32x4*)(ap + 16);
            }
            const int k = kc + kk;
            for (int tc = 0; tc < 4; ++tc) {
                b[tc].q[0] = *(const u32x4*)(bptr[tc] + k);
                b[tc].q[1] = *(const u32x4*)(bptr[tc] + k + 16);
            }
            for (int tr = 0; tr < 2; ++tr)
                for (int tc = 0; tc < 4; ++tc)
                    acc[tr][tc] = __builtin_amdgcn_wmma_f32_16x16x32_bf16(
                        false, a[tr].v, false, b[tc].v,
                        (short)0, acc[tr][tc], false, false);
        }
        __syncthreads();            // buffer may be overwritten next round
    }

    // D layout: VGPR i -> M = tileRow + i + (lane<16 ? 0 : 8), N = tileCol+l15
    const int rAdd = (lane < 16) ? 0 : 8;
    for (int tr = 0; tr < 2; ++tr) {
        for (int tc = 0; tc < 4; ++tc) {
            const int col = colTile0 + tc * 16 + l15;
            const float bv = bias ? bias[col] : 0.0f;
            for (int i = 0; i < 8; ++i) {
                const int row = rowBlock + waveM * 32 + tr * 16 + rAdd + i;
                if (row < M)
                    C[(size_t)row * N + col] = acc[tr][tc][i] + bv;
            }
        }
    }
}

// ---------------------------------------------------------------------------
// LSTM recurrence (one layer): single workgroup, 32 waves, T sequential steps.
// gin: [8, T, 2048] fp32 precomputed input contributions (+biases).
// whh: [2048, 512] bf16.  xout: [8, T, 512] bf16 hidden states.
// h kept as zero-padded 16x512 bf16 tile in LDS (WMMA A matrix), c in regs.
// ---------------------------------------------------------------------------
__global__ __launch_bounds__(1024) void lstm_rec_kernel(
    const float* __restrict__ gin, const __bf16* __restrict__ whh,
    __bf16* __restrict__ xout, int T)
{
    __shared__ __bf16 hsh[16 * 512];     // 16 KB, rows 8..15 stay zero
    __shared__ float  gsh[8 * 2048];     // 64 KB gate staging

    const int tid  = threadIdx.x;
    const int lane = tid & 31;
    const int wave = tid >> 5;
    const int l15  = lane & 15;

    for (int i = tid; i < 16 * 512; i += 1024) hsh[i] = (__bf16)0.0f;
    float c[4] = {0.f, 0.f, 0.f, 0.f};
    __syncthreads();

    const int akoff = (lane < 16) ? 0 : 8;
    const int bkoff = (lane < 16) ? 0 : 8;

    for (int t = 0; t < T; ++t) {
        // ---- gate GEMM: gates[16,2048] = h[16,512] * whh^T ----
        for (int ct4 = 0; ct4 < 4; ++ct4) {
            const int ct      = wave * 4 + ct4;   // 0..127 column tiles
            const int colBase = ct * 16;
            const __bf16* wrow = whh + (size_t)(colBase + l15) * 512 + bkoff;

            v8f acc = {};
            for (int k = 0; k < 512; k += 32) {
                ABfrag a, b;
                a.q[0] = *(const u32x4*)(&hsh[l15 * 512 + k + akoff]);
                a.q[1] = *(const u32x4*)(&hsh[l15 * 512 + k + akoff + 16]);
                b.q[0] = *(const u32x4*)(wrow + k);
                b.q[1] = *(const u32x4*)(wrow + k + 16);
                acc = __builtin_amdgcn_wmma_f32_16x16x32_bf16(
                    false, a.v, false, b.v, (short)0, acc, false, false);
            }
            if (lane < 16) {                      // rows 0..7 (batch) valid
                for (int i = 0; i < 8; ++i)
                    gsh[i * 2048 + colBase + lane] = acc[i];
            }
        }
        __syncthreads();

        // ---- elementwise cell update: 8*512 = 4096 states, 4 per thread ----
        for (int s = 0; s < 4; ++s) {
            const int idx = tid + s * 1024;
            const int n = idx >> 9;
            const int j = idx & 511;
            const float* gb = gin + ((size_t)n * T + t) * 2048;
            const float gi = gsh[n * 2048 + j]        + gb[j];
            const float gf = gsh[n * 2048 + 512 + j]  + gb[512 + j];
            const float gg = gsh[n * 2048 + 1024 + j] + gb[1024 + j];
            const float go = gsh[n * 2048 + 1536 + j] + gb[1536 + j];
            const float cc = sigm(gf) * c[s] + sigm(gi) * tanhf(gg);
            c[s] = cc;
            const float h = sigm(go) * tanhf(cc);
            hsh[n * 512 + j] = (__bf16)h;
            xout[((size_t)n * T + t) * 512 + j] = (__bf16)h;
        }
        __syncthreads();
    }
}

// ---------------------------------------------------------------------------
// Small helper kernels
// ---------------------------------------------------------------------------
__global__ void cvt_bf16_kernel(const float* __restrict__ src,
                                __bf16* __restrict__ dst, size_t n)
{
    size_t i = (size_t)blockIdx.x * blockDim.x + threadIdx.x;
    if (i < n) dst[i] = (__bf16)src[i];
}

__global__ void biassum_kernel(const float* __restrict__ a,
                               const float* __restrict__ b,
                               float* __restrict__ out, int n)
{
    int i = blockIdx.x * blockDim.x + threadIdx.x;
    if (i < n) out[i] = a[i] + b[i];
}

// embedding gather -> bf16; one block per (n,u) row
__global__ void embed_kernel(const int* __restrict__ tgt,
                             const float* __restrict__ embed,
                             __bf16* __restrict__ x0)
{
    const int r = blockIdx.x;                 // 0..N*TO1-1
    const int v = tgt[r];
    const float* e = embed + (size_t)v * E_E;
    __bf16* xp = x0 + (size_t)r * E_E;
    for (int j = threadIdx.x; j < E_E; j += blockDim.x)
        xp[j] = (__bf16)e[j];
}

// A = tanh(enc_p[n,t,:] + dec_p[n,u,:]) -> bf16 (packed dword stores)
__global__ void joint_a_kernel(const float* __restrict__ enc_p,
                               const float* __restrict__ dec_p,
                               __bf16* __restrict__ Ab)
{
    const int r   = blockIdx.x;               // 0..N*TI*TO1-1
    const int n   = r / (T_I * T_O1);
    const int rem = r % (T_I * T_O1);
    const int t   = rem / T_O1;
    const int u   = rem % T_O1;
    const float* ep = enc_p + ((size_t)n * T_I + t) * J_J;
    const float* dp = dec_p + ((size_t)n * T_O1 + u) * J_J;
    unsigned* ap = (unsigned*)(Ab + (size_t)r * J_J);

    const int j2 = threadIdx.x;               // 256 threads, 2 elems each
    union { __bf16 h[2]; unsigned u32; } pk;
    pk.h[0] = (__bf16)tanhf(ep[2 * j2]     + dp[2 * j2]);
    pk.h[1] = (__bf16)tanhf(ep[2 * j2 + 1] + dp[2 * j2 + 1]);
    ap[j2] = pk.u32;
}

// ---------------------------------------------------------------------------
extern "C" void kernel_launch(void* const* d_in, const int* in_sizes, int n_in,
                              void* d_out, int out_size, void* d_ws, size_t ws_size,
                              hipStream_t stream)
{
    const float* enc_out = (const float*)d_in[0];   // [8,200,512]
    const int*   tgt     = (const int*)  d_in[1];   // [8,51]
    const float* embed   = (const float*)d_in[2];   // [1024,512]
    const float* w_ih    = (const float*)d_in[3];   // [3,2048,512]
    const float* w_hh    = (const float*)d_in[4];   // [3,2048,512]
    const float* b_ih    = (const float*)d_in[5];   // [3,2048]
    const float* b_hh    = (const float*)d_in[6];   // [3,2048]
    const float* enc_w   = (const float*)d_in[7];   // [512,512]
    const float* dec_w   = (const float*)d_in[8];   // [512,512]
    const float* dec_b   = (const float*)d_in[9];   // [512]
    const float* out_w   = (const float*)d_in[10];  // [1024,512]
    float* out = (float*)d_out;                     // [8,200,51,1024]

    const int RD = N_B * T_O1;                      // 408 decoder rows
    const int RJ = N_B * T_I * T_O1;                // 81600 joint rows

    // ---- workspace carve-up ----
    char* w = (char*)d_ws;
    auto alloc = [&](size_t bytes) -> char* {
        char* p = w; w += (bytes + 255) & ~(size_t)255; return p;
    };
    __bf16* xA    = (__bf16*)alloc((size_t)RD * E_E * 2);
    __bf16* xB    = (__bf16*)alloc((size_t)RD * E_E * 2);
    __bf16* wihb  = (__bf16*)alloc((size_t)L_L * 4 * H_H * E_E * 2);
    __bf16* whhb  = (__bf16*)alloc((size_t)L_L * 4 * H_H * H_H * 2);
    float*  bsum  = (float*) alloc((size_t)L_L * 4 * H_H * 4);
    float*  gin   = (float*) alloc((size_t)RD * 4 * H_H * 4);
    __bf16* encob = (__bf16*)alloc((size_t)N_B * T_I * 512 * 2);
    __bf16* encwb = (__bf16*)alloc((size_t)J_J * 512 * 2);
    __bf16* decwb = (__bf16*)alloc((size_t)J_J * H_H * 2);
    __bf16* outwb = (__bf16*)alloc((size_t)V_V * J_J * 2);
    float*  enc_p = (float*) alloc((size_t)N_B * T_I * J_J * 4);
    float*  dec_p = (float*) alloc((size_t)RD * J_J * 4);
    __bf16* Ab    = (__bf16*)alloc((size_t)RJ * J_J * 2);

    auto cvt = [&](const float* s, __bf16* d, size_t n) {
        cvt_bf16_kernel<<<dim3((unsigned)((n + 255) / 256)), 256, 0, stream>>>(s, d, n);
    };

    // ---- precision conversions ----
    cvt(enc_out, encob, (size_t)N_B * T_I * 512);
    cvt(w_ih,  wihb, (size_t)L_L * 4 * H_H * E_E);
    cvt(w_hh,  whhb, (size_t)L_L * 4 * H_H * H_H);
    cvt(enc_w, encwb, (size_t)J_J * 512);
    cvt(dec_w, decwb, (size_t)J_J * H_H);
    cvt(out_w, outwb, (size_t)V_V * J_J);
    biassum_kernel<<<dim3((L_L * 4 * H_H + 255) / 256), 256, 0, stream>>>(
        b_ih, b_hh, bsum, L_L * 4 * H_H);

    // ---- embedding lookup ----
    embed_kernel<<<dim3(RD), 256, 0, stream>>>(tgt, embed, xA);

    // ---- 3-layer LSTM ----
    __bf16* xcur = xA;
    __bf16* xnxt = xB;
    for (int l = 0; l < L_L; ++l) {
        // gin = x * w_ih^T + (b_ih + b_hh)    [408 x 2048]
        gemm_bf16_kernel<<<dim3((RD + 127) / 128, (4 * H_H) / 128), 256, 0, stream>>>(
            xcur, wihb + (size_t)l * 4 * H_H * E_E, bsum + (size_t)l * 4 * H_H,
            gin, RD, 4 * H_H, E_E);
        // sequential recurrence
        lstm_rec_kernel<<<dim3(1), 1024, 0, stream>>>(
            gin, whhb + (size_t)l * 4 * H_H * H_H, xnxt, T_O1);
        __bf16* tmp = xcur; xcur = xnxt; xnxt = tmp;
    }

    // ---- projections ----
    gemm_bf16_kernel<<<dim3((N_B * T_I + 127) / 128, J_J / 128), 256, 0, stream>>>(
        encob, encwb, nullptr, enc_p, N_B * T_I, J_J, 512);
    gemm_bf16_kernel<<<dim3((RD + 127) / 128, J_J / 128), 256, 0, stream>>>(
        xcur, decwb, dec_b, dec_p, RD, J_J, H_H);

    // ---- joint: A = tanh(enc_p + dec_p) (bf16), then A * out_w^T ----
    joint_a_kernel<<<dim3(RJ), 256, 0, stream>>>(enc_p, dec_p, Ab);
    gemm_bf16_kernel<<<dim3((RJ + 127) / 128, V_V / 128), 256, 0, stream>>>(
        Ab, outwb, nullptr, out, RJ, V_V, J_J);
}
